// Model_46454366273772
// MI455X (gfx1250) — matrix operand order repaired and checked by reading, not compiled, not run on hardware
//
#include <hip/hip_runtime.h>
#include <math.h>
#include <stdint.h>

typedef _Float16 v16h __attribute__((ext_vector_type(16)));
typedef _Float16 v8h  __attribute__((ext_vector_type(8)));
typedef float    v8f  __attribute__((ext_vector_type(8)));

#define SHUF16(lo,hi) __builtin_shufflevector((lo),(hi),0,1,2,3,4,5,6,7,8,9,10,11,12,13,14,15)

namespace {
constexpr int B_    = 8;
constexpr int T_    = 256;
constexpr int E_    = 64;
constexpr int H_    = 512;
constexpr int V_    = 50257;
constexpr int DK_   = 128;     // head dim
constexpr int H2_   = 1024;    // 2H
constexpr int KG_   = 1536;    // 2H + H  (gate GEMM K)
constexpr int NWG_  = 16;      // workgroups in the recurrent scan
constexpr int NTT_  = 3144;    // ceil(50257/64)*4 : N tiles of 16 (padded to 50304)
constexpr int VE_   = V_ + E_; // 50321
constexpr int ROWS_ = B_ * T_; // 2048
}

__device__ __forceinline__ float sigmf_(float x) { return 1.0f / (1.0f + expf(-x)); }

__device__ __forceinline__ v8f vzero8() {
  v8f z;
#pragma unroll
  for (int i = 0; i < 8; ++i) z[i] = 0.0f;
  return z;
}

// ---------------- grid barrier (sense reversing) ----------------
__device__ __forceinline__ void grid_barrier(int* bar) {
  __threadfence();
  __syncthreads();
  if (threadIdx.x == 0) {
    int g   = __hip_atomic_load(bar + 1, __ATOMIC_ACQUIRE, __HIP_MEMORY_SCOPE_AGENT);
    int old = __hip_atomic_fetch_add(bar, 1, __ATOMIC_ACQ_REL, __HIP_MEMORY_SCOPE_AGENT);
    if (old == NWG_ - 1) {
      __hip_atomic_store(bar, 0, __ATOMIC_RELAXED, __HIP_MEMORY_SCOPE_AGENT);
      __hip_atomic_fetch_add(bar + 1, 1, __ATOMIC_ACQ_REL, __HIP_MEMORY_SCOPE_AGENT);
    } else {
      while (__hip_atomic_load(bar + 1, __ATOMIC_ACQUIRE, __HIP_MEMORY_SCOPE_AGENT) == g)
        __builtin_amdgcn_s_sleep(2);
    }
  }
  __syncthreads();
}

// ---------------- init: hx0 = mean(ents), a0 = 0, zero barrier ----------------
__global__ __launch_bounds__(256) void k_init(const float* __restrict__ ents,
                                              float* __restrict__ hxG, float* __restrict__ aG,
                                              int* __restrict__ bar) {
  int i = blockIdx.x * 256 + threadIdx.x;          // 4096 threads
  if (i < B_ * H_) {
    int b = i >> 9, h = i & 511;
    float s = 0.f;
    for (int e = 0; e < E_; ++e) s += ents[(size_t)(b * E_ + e) * H_ + h];
    hxG[i] = s * (1.0f / E_);   // buffer 0
    aG[i]  = 0.0f;              // buffer 0
  }
  if (i < 8) bar[i] = 0;
}

// ---------------- pack gate weights: [K=1536][N=2048] -> f16 WMMA-B swizzle ----------------
// per-WG n permutation: WG j owns cols c in [0,128): gate g=c>>5, hid=j*32+(c&31)
__global__ __launch_bounds__(256) void k_pack_gates(const float* __restrict__ w_ih,
                                                    const float* __restrict__ w_hh,
                                                    _Float16* __restrict__ wg) {
  int i = blockIdx.x * 256 + threadIdx.x;
  if (i >= NWG_ * 8 * 48 * 512) return;
  int q = i >> 9, r = i & 511;
  int lane = r >> 4, e = r & 15;
  int kk = q % 48; int rem = q / 48;
  int tt = rem & 7; int j = rem >> 3;
  int k = kk * 32 + ((lane >> 4) * 16) + e;        // 0..1535
  int c = tt * 16 + (lane & 15);                   // 0..127 local col
  int g = c >> 5, hid = j * 32 + (c & 31);
  int n = g * H_ + hid;                            // row of w_ih/w_hh
  float v = (k < H2_) ? w_ih[(size_t)n * H2_ + k] : w_hh[(size_t)n * H_ + (k - H2_)];
  wg[i] = (_Float16)v;
}

// ---------------- pack out_w: [K=1024][N padded 50304] -> f16 WMMA-B swizzle ----------------
__global__ __launch_bounds__(256) void k_pack_outw(const float* __restrict__ out_w,
                                                   _Float16* __restrict__ wpk) {
  size_t i = (size_t)blockIdx.x * 256 + threadIdx.x;
  if (i >= (size_t)NTT_ * 32 * 512) return;
  size_t q = i >> 9; int r = (int)(i & 511);
  int lane = r >> 4, e = r & 15;
  int kk = (int)(q & 31); int nt = (int)(q >> 5);
  int k = kk * 32 + ((lane >> 4) * 16) + e;        // 0..1023
  int n = nt * 16 + (lane & 15);
  float v = (n < V_) ? out_w[(size_t)k * V_ + n] : 0.0f;
  wpk[i] = (_Float16)v;
}

// ---------------- embedding gather -> f16 [T][B][H] ----------------
__global__ __launch_bounds__(256) void k_emb(const int* __restrict__ outp,
                                             const float* __restrict__ embed_w,
                                             _Float16* __restrict__ embh) {
  int i = blockIdx.x * 256 + threadIdx.x;          // T*B*H = 1,048,576
  if (i >= T_ * B_ * H_) return;
  int h = i & 511; int b = (i >> 9) & 7; int t = i >> 12;
  int idx = outp[b * T_ + t];
  embh[i] = (_Float16)embed_w[(size_t)idx * H_ + h];
}

// ---------------- K,V precompute: [B][E][NH*DK] ----------------
__global__ __launch_bounds__(256) void k_kv(const float* __restrict__ ents,
                                            const float* __restrict__ wk, const float* __restrict__ bk,
                                            const float* __restrict__ wv, const float* __restrict__ bv,
                                            float* __restrict__ kbuf, float* __restrict__ vbuf) {
  int i = blockIdx.x * 256 + threadIdx.x;          // B*E*H = 262,144
  if (i >= B_ * E_ * H_) return;
  int n = i & 511; int be = i >> 9;                // be = b*E+e
  const float* ep = ents + (size_t)be * H_;
  float sk = bk[n], sv = bv[n];
  for (int k = 0; k < H_; ++k) {
    float x = ep[k];
    sk += x * wk[(size_t)k * H_ + n];
    sv += x * wv[(size_t)k * H_ + n];
  }
  kbuf[i] = sk; vbuf[i] = sv;
}

// ---------------- recurrent scan: 16 WGs x 256 thr, WMMA gates + attention ----------------
__global__ __launch_bounds__(256)
void k_scan(const _Float16* __restrict__ wg, const _Float16* __restrict__ embh,
            const float* __restrict__ b_ih, const float* __restrict__ b_hh,
            const float* __restrict__ wq, const float* __restrict__ bq,
            const float* __restrict__ wo, const float* __restrict__ bo,
            const float* __restrict__ kbuf, const float* __restrict__ vbuf,
            const int* __restrict__ ent_lens,
            float* __restrict__ hxG, float* __restrict__ aG,   // double buffered [2][B*H]
            float* __restrict__ ctxG, float* __restrict__ lbuf,
            int* __restrict__ bar) {
  __shared__ __align__(16) _Float16 Xs[16 * KG_];  // A matrix 16x1536 (rows 8..15 zero)
  __shared__ float gS[8 * 128];
  __shared__ float cxS[8 * 32];
  __shared__ float qS[2 * 128];
  __shared__ float scS[2 * 64];
  __shared__ float redS[4];

  const int tid = threadIdx.x;
  const int lane = tid & 31;
  const int wave = tid >> 5;
  const int j = blockIdx.x;

  // zero pad rows (m = 8..15)
  for (int i = tid; i < 8 * KG_; i += 256) Xs[8 * KG_ + i] = (_Float16)0.0f;
  // cx0 = hx0 (slice owned by this WG)
  { int b = tid >> 5, hl = tid & 31; cxS[tid] = hxG[b * H_ + j * 32 + hl]; }
  __syncthreads();

  for (int t = 0; t < T_; ++t) {
    const float* hr = hxG + (size_t)(t & 1) * (B_ * H_);        // h_{t-1}
    const float* ar = aG  + (size_t)(t & 1) * (B_ * H_);        // a_{t-1}
    float* hw = hxG + (size_t)((t + 1) & 1) * (B_ * H_);        // h_t
    float* aw = aG  + (size_t)((t + 1) & 1) * (B_ * H_);        // a_t

    // build X = [a_{t-1} | emb_t | h_{t-1}] rows 0..7, f16
    for (int i = tid; i < 8 * KG_; i += 256) {
      int m = i / KG_, k = i - m * KG_;
      if (k < H_)            Xs[i] = (_Float16)ar[m * H_ + k];
      else if (k < H2_)      Xs[i] = embh[(size_t)(t * B_ + m) * H_ + (k - H_)];
      else                   Xs[i] = (_Float16)hr[m * H_ + (k - H2_)];
    }
    __syncthreads();

    // gates GEMM: this wave owns n-tile 'wave' of the WG's 128 columns
    v8f acc = vzero8();
    for (int kk = 0; kk < 48; ++kk) {
      const _Float16* xr = &Xs[(lane & 15) * KG_ + kk * 32 + ((lane >> 4) * 16)];
      v8h alo = *(const v8h*)xr;
      v8h ahi = *(const v8h*)(xr + 8);
      v16h af = SHUF16(alo, ahi);
      const _Float16* bp = wg + ((size_t)((j * 8 + wave) * 48 + kk)) * 512 + lane * 16;
      v8h blo = *(const v8h*)bp;
      v8h bhi = *(const v8h*)(bp + 8);
      v16h bf = SHUF16(blo, bhi);
      acc = __builtin_amdgcn_wmma_f32_16x16x32_f16(false, af, false, bf, (short)0, acc,
                                                   false, false);
    }
    if (lane < 16) {
#pragma unroll
      for (int r = 0; r < 8; ++r) gS[r * 128 + wave * 16 + lane] = acc[r];
    }
    __syncthreads();

    // LSTM activation for this WG's 32 hidden units x 8 batch
    {
      int b = tid >> 5, hl = tid & 31;
      int nh = j * 32 + hl;
      float i_ = gS[b * 128 +       hl] + b_ih[          nh] + b_hh[          nh];
      float f_ = gS[b * 128 +  32 + hl] + b_ih[H_      + nh] + b_hh[H_      + nh];
      float g_ = gS[b * 128 +  64 + hl] + b_ih[2 * H_  + nh] + b_hh[2 * H_  + nh];
      float o_ = gS[b * 128 +  96 + hl] + b_ih[3 * H_  + nh] + b_hh[3 * H_  + nh];
      float c  = sigmf_(f_) * cxS[tid] + sigmf_(i_) * tanhf(g_);
      cxS[tid] = c;
      float h  = sigmf_(o_) * tanhf(c);
      hw[b * H_ + nh] = h;
      lbuf[((size_t)(b * T_ + t)) * H2_ + nh] = h;
    }
    grid_barrier(bar);

    // attention: WG j handles (b,h) pairs 2j and 2j+1
    {
      int hv  = tid >> 7;                 // which pair (0/1)
      int loc = tid & 127;
      int p = j * 2 + hv;
      int b = p >> 2, h = p & 3;
      // q = hx_t @ wq + bq  (this pair's 128 cols)
      int n = h * DK_ + loc;
      float qv = bq[n];
      for (int k = 0; k < H_; ++k) qv += hw[b * H_ + k] * wq[(size_t)k * H_ + n];
      qS[hv * 128 + loc] = qv;
      __syncthreads();
      if (loc < 64) {
        int e = loc;
        float sc = -1e9f;
        if (e <= ent_lens[b]) {
          float s = 0.f;
          const float* kp = kbuf + ((size_t)(b * E_ + e)) * H_ + h * DK_;
          for (int d = 0; d < DK_; ++d) s += qS[hv * 128 + d] * kp[d];
          sc = s * 0.08838834764831845f;  // 1/sqrt(128)
        }
        scS[hv * 64 + e] = sc;
      }
      __syncthreads();
      if (loc == 0) {
        float mx = -1e30f;
        for (int e = 0; e < 64; ++e) mx = fmaxf(mx, scS[hv * 64 + e]);
        float sm = 0.f;
        for (int e = 0; e < 64; ++e) sm += expf(scS[hv * 64 + e] - mx);
        redS[hv * 2] = mx; redS[hv * 2 + 1] = sm;
      }
      __syncthreads();
      {
        float mx = redS[hv * 2], inv = 1.0f / redS[hv * 2 + 1];
        float cv = 0.f;
        const float* vp = vbuf + ((size_t)(b * E_)) * H_ + h * DK_ + loc;
        for (int e = 0; e < 64; ++e) cv += expf(scS[hv * 64 + e] - mx) * inv * vp[(size_t)e * H_];
        ctxG[b * H_ + h * DK_ + loc] = cv;
      }
    }
    grid_barrier(bar);

    // a_t = ctx @ wo + bo (this WG's 32 cols)
    {
      int b = tid >> 5, hl = tid & 31;
      int n = j * 32 + hl;
      float av = bo[n];
      for (int k = 0; k < H_; ++k) av += ctxG[b * H_ + k] * wo[(size_t)k * H_ + n];
      aw[b * H_ + n] = av;
      lbuf[((size_t)(b * T_ + t)) * H2_ + H_ + n] = av;
    }
    grid_barrier(bar);
    __syncthreads();
  }
}

// ---------------- l -> f16 ----------------
__global__ __launch_bounds__(256) void k_l2h(const float* __restrict__ l, _Float16* __restrict__ lh) {
  int i = blockIdx.x * 256 + threadIdx.x;
  if (i < ROWS_ * H2_) lh[i] = (_Float16)l[i];
}

// ---------------- switch gate: s = sigmoid(l @ switch_w + b) ----------------
__global__ __launch_bounds__(256) void k_switch(const float* __restrict__ l,
                                                const float* __restrict__ sw,
                                                const float* __restrict__ sb,
                                                float* __restrict__ sbuf) {
  int row = blockIdx.x * 256 + threadIdx.x;
  if (row >= ROWS_) return;
  const float* lp = l + (size_t)row * H2_;
  float s = sb[0];
  for (int k = 0; k < H2_; ++k) s += lp[k] * sw[k];
  sbuf[row] = sigmf_(s);
}

// ---------------- m = l @ mat_w + mat_b ----------------
__global__ __launch_bounds__(256) void k_matw(const float* __restrict__ l,
                                              const float* __restrict__ mw,
                                              const float* __restrict__ mb,
                                              float* __restrict__ mbuf) {
  int i = blockIdx.x * 256 + threadIdx.x;          // ROWS*H = 1,048,576
  if (i >= ROWS_ * H_) return;
  int n = i & 511; int row = i >> 9;
  const float* lp = l + (size_t)row * H2_;
  float s = mb[n];
  for (int k = 0; k < H2_; ++k) s += lp[k] * mw[(size_t)k * H_ + n];
  mbuf[i] = s;
}

// ---------------- big WMMA GEMM: logits = l_f16 @ out_w_f16 + out_b -> d_out ----------------
// B tiles (shared by all 8 waves) staged in LDS via async global->LDS copies (ASYNCcnt),
// double-buffered so stage kk+1 is in flight while stage kk feeds the WMMAs.
__global__ __launch_bounds__(256)
void k_outgemm(const _Float16* __restrict__ lh, const _Float16* __restrict__ wpk,
               const float* __restrict__ out_b, float* __restrict__ out) {
  __shared__ __align__(16) _Float16 Bs[2][4 * 512];   // 2 x 4KB
  const int tid = threadIdx.x, lane = tid & 31, wave = tid >> 5;
  const int mbase = blockIdx.x * 128 + wave * 16;  // wave's 16 rows
  const int nt0 = blockIdx.y * 4;                  // 4 n-tiles (N=64)

  // async staging addresses: thread tid copies 16B; tile ti = tid>>6
  const int ti = tid >> 6;
  const uint64_t gthr = (uint64_t)(uintptr_t)wpk +
                        ((uint64_t)(nt0 + ti) * 32) * 1024 + (uint64_t)(tid & 63) * 16;
  const unsigned lds0 = (unsigned)(uintptr_t)(&Bs[0][0]) + (unsigned)tid * 16;
  const unsigned lds1 = (unsigned)(uintptr_t)(&Bs[1][0]) + (unsigned)tid * 16;

  v8f acc[4];
#pragma unroll
  for (int i = 0; i < 4; ++i) acc[i] = vzero8();
  const int arow = mbase + (lane & 15);
  const _Float16* abase = lh + (size_t)arow * H2_ + ((lane >> 4) * 16);

  // prologue: stage kk=0 into buffer 0
  {
    uint64_t ga = gthr;
    asm volatile("global_load_async_to_lds_b128 %0, %1, off"
                 :: "v"(lds0), "v"(ga) : "memory");
  }
  for (int kk = 0; kk < 32; ++kk) {
    if (kk + 1 < 32) {
      uint64_t ga = gthr + (uint64_t)(kk + 1) * 1024;
      unsigned la = ((kk + 1) & 1) ? lds1 : lds0;
      asm volatile("global_load_async_to_lds_b128 %0, %1, off"
                   :: "v"(la), "v"(ga) : "memory");
      asm volatile("s_wait_asynccnt 0x1" ::: "memory");   // stage kk arrived (1 in flight)
    } else {
      asm volatile("s_wait_asynccnt 0x0" ::: "memory");
    }
    __syncthreads();                                      // all waves' chunks present

    const _Float16* ap = abase + kk * 32;
    v8h alo = *(const v8h*)ap;
    v8h ahi = *(const v8h*)(ap + 8);
    v16h af = SHUF16(alo, ahi);
    const _Float16* bb = &Bs[kk & 1][0];
#pragma unroll
    for (int i = 0; i < 4; ++i) {
      const _Float16* bp = bb + i * 512 + lane * 16;
      v8h blo = *(const v8h*)bp;
      v8h bhi = *(const v8h*)(bp + 8);
      v16h bf = SHUF16(blo, bhi);
      acc[i] = __builtin_amdgcn_wmma_f32_16x16x32_f16(false, af, false, bf, (short)0, acc[i],
                                                      false, false);
    }
    __syncthreads();                                      // done reading buf before re-stage
  }

  const int mrow = mbase + ((lane < 16) ? 0 : 8);
#pragma unroll
  for (int i = 0; i < 4; ++i) {
    int n = (nt0 + i) * 16 + (lane & 15);
    if (n >= V_) continue;
    float bn = out_b[n];
#pragma unroll
    for (int r = 0; r < 8; ++r)
      out[(size_t)(mrow + r) * VE_ + n] = acc[i][r] + bn;
  }
}

// ---------------- per-row finalize: softmax*s, log, E-side copy dist ----------------
__global__ __launch_bounds__(256)
void k_final(const float* __restrict__ sbuf, const float* __restrict__ mbuf,
             const float* __restrict__ ents, const int* __restrict__ ent_lens,
             float* __restrict__ out, float* __restrict__ zout) {
  __shared__ float red[256];
  __shared__ float ms[H_];
  __shared__ float scS[E_];
  __shared__ float st[2];
  const int row = blockIdx.x, tid = threadIdx.x;
  const int b = row / T_;
  const size_t base = (size_t)row * VE_;

  float mx = -1e30f;
  for (int v = tid; v < V_; v += 256) mx = fmaxf(mx, out[base + v]);
  red[tid] = mx; __syncthreads();
  for (int s = 128; s > 0; s >>= 1) { if (tid < s) red[tid] = fmaxf(red[tid], red[tid + s]); __syncthreads(); }
  mx = red[0]; __syncthreads();

  float sm = 0.f;
  for (int v = tid; v < V_; v += 256) sm += expf(out[base + v] - mx);
  red[tid] = sm; __syncthreads();
  for (int s = 128; s > 0; s >>= 1) { if (tid < s) red[tid] += red[tid + s]; __syncthreads(); }
  sm = red[0]; __syncthreads();

  const float sw = sbuf[row];
  const float scale = sw / sm;
  for (int v = tid; v < V_; v += 256)
    out[base + v] = logf(expf(out[base + v] - mx) * scale + 1e-6f);

  // entity distribution
  for (int k = tid; k < H_; k += 256) ms[k] = mbuf[(size_t)row * H_ + k];
  __syncthreads();
  if (tid < E_) {
    float sc = -1e9f;
    if (tid <= ent_lens[b]) {
      float s = 0.f;
      const float* ep = ents + ((size_t)(b * E_ + tid)) * H_;
      for (int k = 0; k < H_; ++k) s += ms[k] * ep[k];
      sc = s;
    }
    scS[tid] = sc;
  }
  __syncthreads();
  if (tid == 0) {
    float m2 = -1e30f;
    for (int e = 0; e < E_; ++e) m2 = fmaxf(m2, scS[e]);
    float s2 = 0.f;
    for (int e = 0; e < E_; ++e) s2 += expf(scS[e] - m2);
    st[0] = m2; st[1] = s2;
  }
  __syncthreads();
  if (tid < E_) {
    float z = expf(scS[tid] - st[0]) / st[1] * (1.0f - sw);
    out[base + V_ + tid] = logf(z + 1e-6f);
    zout[(size_t)row * E_ + tid] = z;
  }
}

// ---------------- host ----------------
extern "C" void kernel_launch(void* const* d_in, const int* in_sizes, int n_in,
                              void* d_out, int out_size, void* d_ws, size_t ws_size,
                              hipStream_t stream) {
  (void)in_sizes; (void)n_in; (void)out_size; (void)ws_size;
  const int*   outp     = (const int*)  d_in[0];
  const float* ents     = (const float*)d_in[1];
  const int*   ent_lens = (const int*)  d_in[2];
  const float* embed_w  = (const float*)d_in[3];
  const float* w_ih     = (const float*)d_in[4];
  const float* b_ih     = (const float*)d_in[5];
  const float* w_hh     = (const float*)d_in[6];
  const float* b_hh     = (const float*)d_in[7];
  const float* wq = (const float*)d_in[8];  const float* bq = (const float*)d_in[9];
  const float* wk = (const float*)d_in[10]; const float* bk = (const float*)d_in[11];
  const float* wv = (const float*)d_in[12]; const float* bv = (const float*)d_in[13];
  const float* wo = (const float*)d_in[14]; const float* bo = (const float*)d_in[15];
  const float* switch_w = (const float*)d_in[16];
  const float* switch_b = (const float*)d_in[17];
  const float* out_w    = (const float*)d_in[18];
  const float* out_b    = (const float*)d_in[19];
  const float* mat_w    = (const float*)d_in[20];
  const float* mat_b    = (const float*)d_in[21];

  char* ws = (char*)d_ws;
  constexpr size_t OFF_BAR = 0;
  constexpr size_t OFF_WG  = 256;
  constexpr size_t SZ_WG   = (size_t)NWG_ * 8 * 48 * 512 * 2;          // 6 MB
  constexpr size_t OFF_WO  = OFF_WG + SZ_WG;
  constexpr size_t SZ_WO   = (size_t)NTT_ * 32 * 512 * 2;              // 103 MB
  constexpr size_t OFF_L   = OFF_WO + SZ_WO;
  constexpr size_t SZ_L    = (size_t)ROWS_ * H2_ * 4;                  // 8 MB
  constexpr size_t OFF_LH  = OFF_L + SZ_L;
  constexpr size_t SZ_LH   = (size_t)ROWS_ * H2_ * 2;                  // 4 MB
  constexpr size_t OFF_EMB = OFF_LH + SZ_LH;
  constexpr size_t SZ_EMB  = (size_t)T_ * B_ * H_ * 2;                 // 2 MB
  constexpr size_t OFF_HX  = OFF_EMB + SZ_EMB;
  constexpr size_t SZ_HX   = (size_t)2 * B_ * H_ * 4;                  // double buffered
  constexpr size_t OFF_A   = OFF_HX + SZ_HX;
  constexpr size_t SZ_A    = (size_t)2 * B_ * H_ * 4;
  constexpr size_t OFF_CTX = OFF_A + SZ_A;
  constexpr size_t SZ_CTX  = (size_t)B_ * H_ * 4;
  constexpr size_t OFF_KB  = OFF_CTX + SZ_CTX;
  constexpr size_t SZ_KB   = (size_t)B_ * E_ * H_ * 4;                 // 1 MB
  constexpr size_t OFF_VB  = OFF_KB + SZ_KB;
  constexpr size_t SZ_VB   = SZ_KB;
  constexpr size_t OFF_SB  = OFF_VB + SZ_VB;
  constexpr size_t SZ_SB   = (size_t)ROWS_ * 4;
  constexpr size_t OFF_MB  = OFF_SB + SZ_SB;

  int*       bar  = (int*)(ws + OFF_BAR);
  _Float16*  wg   = (_Float16*)(ws + OFF_WG);
  _Float16*  wpk  = (_Float16*)(ws + OFF_WO);
  float*     lbuf = (float*)(ws + OFF_L);
  _Float16*  lh   = (_Float16*)(ws + OFF_LH);
  _Float16*  embh = (_Float16*)(ws + OFF_EMB);
  float*     hxG  = (float*)(ws + OFF_HX);
  float*     aG   = (float*)(ws + OFF_A);
  float*     ctxG = (float*)(ws + OFF_CTX);
  float*     kbuf = (float*)(ws + OFF_KB);
  float*     vbuf = (float*)(ws + OFF_VB);
  float*     sbuf = (float*)(ws + OFF_SB);
  float*     mbuf = (float*)(ws + OFF_MB);

  float* oout = (float*)d_out;
  float* zout = oout + (size_t)ROWS_ * VE_;

  k_init<<<16, 256, 0, stream>>>(ents, hxG, aG, bar);
  k_pack_gates<<<(NWG_ * 8 * 48 * 512 + 255) / 256, 256, 0, stream>>>(w_ih, w_hh, wg);
  {
    size_t tot = (size_t)NTT_ * 32 * 512;
    k_pack_outw<<<(unsigned)((tot + 255) / 256), 256, 0, stream>>>(out_w, wpk);
  }
  k_emb<<<(T_ * B_ * H_ + 255) / 256, 256, 0, stream>>>(outp, embed_w, embh);
  k_kv<<<(B_ * E_ * H_ + 255) / 256, 256, 0, stream>>>(ents, wk, bk, wv, bv, kbuf, vbuf);

  k_scan<<<NWG_, 256, 0, stream>>>(wg, embh, b_ih, b_hh, wq, bq, wo, bo,
                                   kbuf, vbuf, ent_lens, hxG, aG, ctxG, lbuf, bar);

  k_l2h<<<(ROWS_ * H2_ + 255) / 256, 256, 0, stream>>>(lbuf, lh);
  k_switch<<<(ROWS_ + 255) / 256, 256, 0, stream>>>(lbuf, switch_w, switch_b, sbuf);
  k_matw<<<(ROWS_ * H_ + 255) / 256, 256, 0, stream>>>(lbuf, mat_w, mat_b, mbuf);

  dim3 ggrid(ROWS_ / 128, NTT_ / 4);   // (16, 786)
  k_outgemm<<<ggrid, 256, 0, stream>>>(lh, wpk, out_b, oout);

  k_final<<<ROWS_, 256, 0, stream>>>(sbuf, mbuf, ents, ent_lens, oout, zout);
}
